// TdLstm_27925877359074
// MI455X (gfx1250) — matrix operand order repaired and checked by reading, not compiled, not run on hardware
//
#include <hip/hip_runtime.h>
#include <hip/hip_bf16.h>

// ---------------- problem constants ----------------
#define VOCAB 50000
#define EMB   300
#define HID   512
#define TSEQ  128
#define BATCH 256
#define NCLS  3
#define GATES (4*HID)          // 2048
#define KDIM  (EMB+HID)        // 812
#define KPAD  832              // 26 * 32
#define KB    (KPAD/32)        // 26 k-blocks of 32 (bf16 WMMA K)
#define NT    (GATES/16)       // 128 N tiles
#define WB_SIDE ((size_t)NT*KB*32*16)   // halfs per side = 1,703,936

typedef __attribute__((ext_vector_type(16))) __bf16 v16bf;
typedef __attribute__((ext_vector_type(8)))  __bf16 v8bf;
typedef __attribute__((ext_vector_type(8)))  float  v8f;

static __device__ __forceinline__ __bf16 f2bf(float f) {
    unsigned u = __builtin_bit_cast(unsigned, f);
    unsigned r = (u + 0x7FFFu + ((u >> 16) & 1u)) >> 16;   // round-nearest-even
    unsigned short s = (unsigned short)r;
    return __builtin_bit_cast(__bf16, s);
}
static __device__ __forceinline__ float bf2f(__bf16 b) {
    unsigned u = ((unsigned)__builtin_bit_cast(unsigned short, b)) << 16;
    return __builtin_bit_cast(float, u);
}
static __device__ __forceinline__ float sigm(float x) {
    return 1.0f / (1.0f + __expf(-x));
}

// ---------------------------------------------------------------------------
// Kernel 1: convert W_l / W_r (f32 [812][2048]) into bf16, zero-padded to
// K=832, in pre-swizzled WMMA-B-fragment order:
//   idx = ((nt*KB + kb)*32 + lane)*16 + e
//   lane<16 : col = nt*16+lane, K = kb*32 + e        (e = 0..15)
//   lane>=16: col = nt*16+lane-16, K = kb*32 + 16+e
// so each lane's 16-half fragment is one contiguous 32-byte global load.
// ---------------------------------------------------------------------------
__global__ __launch_bounds__(256) void prep_weights(const float* __restrict__ Wl,
                                                    const float* __restrict__ Wr,
                                                    __bf16* __restrict__ wb) {
    int idx = blockIdx.x * 256 + threadIdx.x;
    const int per_side = KPAD * GATES;
    if (idx >= 2 * per_side) return;
    int side = idx / per_side;
    int rem  = idx - side * per_side;
    int k    = rem / GATES;
    int col  = rem - k * GATES;
    const float* W = side ? Wr : Wl;
    float v = (k < KDIM) ? W[(size_t)k * GATES + col] : 0.0f;
    int kb = k >> 5, kr = k & 31;
    int nt = col >> 4, nc = col & 15;
    int lane = (kr < 16) ? nc : (16 + nc);
    int e    = kr & 15;
    size_t o = (size_t)side * WB_SIDE +
               ((((size_t)nt * KB + kb) * 32 + lane) * 16 + e);
    wb[o] = f2bf(v);
}

// ---------------------------------------------------------------------------
// Kernel 2: fused LSTM recurrence. One workgroup per (side, 16-row batch
// tile); 8 waves; entire T=128 loop inside the kernel; no inter-WG sync.
// LDS (dynamic): Abuf bf16[16][832] | zbuf f32[16][2048] | cbuf f32[16][512]
//                | bias f32[2048]
// ---------------------------------------------------------------------------
#define SM_A  0
#define SM_Z  (SM_A + 16*KPAD*2)            // 26624
#define SM_C  (SM_Z + 16*GATES*4)           // +131072
#define SM_B  (SM_C + 16*HID*4)             // +32768
#define SM_TOT (SM_B + GATES*4)             // +8192 = 198656 bytes

__global__ __launch_bounds__(256) void lstm_recur(
        const int*   __restrict__ left_ids,  const int* __restrict__ right_ids,
        const int*   __restrict__ left_len,  const int* __restrict__ right_len,
        const float* __restrict__ emb,
        const float* __restrict__ b_l,       const float* __restrict__ b_r,
        const __bf16* __restrict__ wb,
        float* __restrict__ hcat) {
    extern __shared__ char smem[];
    __bf16* Abuf = (__bf16*)(smem + SM_A);
    float*  zbuf = (float*)(smem + SM_Z);
    float*  cbuf = (float*)(smem + SM_C);
    float*  bias = (float*)(smem + SM_B);
    __shared__ int sids[16];
    __shared__ int slen[16];

    const int tid   = threadIdx.x;
    const int side  = blockIdx.x >> 4;       // 0 = left, 1 = right
    const int btile = blockIdx.x & 15;       // 16 batch rows per tile
    const int*   ids  = side ? right_ids : left_ids;
    const int*   lens = side ? right_len : left_len;
    const float* bg   = side ? b_r : b_l;
    const __bf16* W   = wb + (size_t)side * WB_SIDE;

    // ---- init: bias -> LDS, c = 0, h region + K padding of Abuf = 0 ----
    for (int i = tid; i < GATES; i += 256) bias[i] = bg[i];
    for (int i = tid; i < 16 * HID; i += 256) cbuf[i] = 0.0f;
    for (int i = tid; i < 16 * (KPAD - EMB); i += 256) {
        int r = i / (KPAD - EMB), c = i - r * (KPAD - EMB);
        Abuf[r * KPAD + EMB + c] = f2bf(0.0f);
    }
    if (tid < 16) slen[tid] = lens[btile * 16 + tid];
    __syncthreads();

    const int lane  = tid & 31;
    const int wv    = tid >> 5;              // wave id 0..7
    const int arow  = lane & 15;             // A-frag: M row
    const int ahalf = (lane >> 4) << 4;      // 0 or 16 (A layout split)
    const __bf16* wlane = W + (size_t)lane * 16;
    const int colr = lane & 15;              // D layout: N within tile
    const int rb   = (lane < 16) ? 0 : 8;    // D layout: M base

    for (int t = 0; t < TSEQ; ++t) {
        // ---- gather x_t (embedding rows) into Abuf[:, 0:300] as bf16 ----
        if (tid < 16) sids[tid] = ids[(btile * 16 + tid) * TSEQ + t];
        __syncthreads();
        for (int i = tid; i < 16 * EMB; i += 256) {
            int r = i / EMB, e = i - r * EMB;
            Abuf[r * KPAD + e] = f2bf(emb[(size_t)sids[r] * EMB + e]);
        }
        __syncthreads();

        // ---- z[16][2048] = [x_t | h] @ W  via bf16 WMMA, f32 accum ----
        // wave wv owns N tiles [wv*16, wv*16+16), blocked 8 at a time
        for (int g = 0; g < 2; ++g) {
            const int nt0 = wv * 16 + g * 8;
            v8f acc[8];
            const v8f vzero = {0.f,0.f,0.f,0.f,0.f,0.f,0.f,0.f};
#pragma unroll
            for (int j = 0; j < 8; ++j) acc[j] = vzero;

            for (int kb = 0; kb < KB; ++kb) {
                const __bf16* ap = Abuf + arow * KPAD + (kb << 5) + ahalf;
                v8bf a0 = *(const v8bf*)(ap);
                v8bf a1 = *(const v8bf*)(ap + 8);
                v16bf a;
#pragma unroll
                for (int i = 0; i < 8; ++i) { a[i] = a0[i]; a[i + 8] = a1[i]; }
#pragma unroll
                for (int j = 0; j < 8; ++j) {
                    const v16bf bfrag =
                        *(const v16bf*)(wlane + (((size_t)(nt0 + j) * KB + kb) << 9));
                    acc[j] = __builtin_amdgcn_wmma_f32_16x16x32_bf16(
                        false, a, false, bfrag, (short)0, acc[j], false, false);
                }
            }
#pragma unroll
            for (int j = 0; j < 8; ++j) {
                const int col = ((nt0 + j) << 4) + colr;
#pragma unroll
                for (int v = 0; v < 8; ++v)
                    zbuf[(rb + v) * GATES + col] = acc[j][v];
            }
        }
        __syncthreads();

        // ---- gates (TF order i,j,f,o; forget_bias = 1.0) + length mask ----
        for (int i = tid; i < 16 * HID; i += 256) {
            int r = i >> 9, hs = i & (HID - 1);
            if (t < slen[r]) {
                const float* zr = zbuf + r * GATES;
                float zi = zr[hs]            + bias[hs];
                float zj = zr[HID + hs]      + bias[HID + hs];
                float zf = zr[2 * HID + hs]  + bias[2 * HID + hs];
                float zo = zr[3 * HID + hs]  + bias[3 * HID + hs];
                float c  = cbuf[i];
                float cn = c * sigm(zf + 1.0f) + sigm(zi) * tanhf(zj);
                float hn = tanhf(cn) * sigm(zo);
                cbuf[i] = cn;
                Abuf[r * KPAD + EMB + hs] = f2bf(hn);       // feeds next step
                hcat[(size_t)(btile * 16 + r) * (2 * HID) + side * HID + hs] = hn;
            }
        }
        __syncthreads();
    }
}

// ---------------------------------------------------------------------------
// Kernel 3: logits[256,3] = Hcat[256,1024] @ W_d[1024,3] + b_d (tiny)
// ---------------------------------------------------------------------------
__global__ __launch_bounds__(256) void logits_kernel(const float* __restrict__ hcat,
                                                     const float* __restrict__ Wd,
                                                     const float* __restrict__ bd,
                                                     float* __restrict__ out) {
    int idx = blockIdx.x * 256 + threadIdx.x;
    if (idx >= BATCH * NCLS) return;
    int b = idx / NCLS, c = idx - b * NCLS;
    const float* h = hcat + (size_t)b * (2 * HID);
    float s = bd[c];
    for (int k = 0; k < 2 * HID; ++k) s = fmaf(h[k], Wd[k * NCLS + c], s);
    out[idx] = s;
}

// ---------------------------------------------------------------------------
extern "C" void kernel_launch(void* const* d_in, const int* in_sizes, int n_in,
                              void* d_out, int out_size, void* d_ws, size_t ws_size,
                              hipStream_t stream) {
    const int*   left_ids  = (const int*)  d_in[0];
    const int*   right_ids = (const int*)  d_in[1];
    const int*   left_len  = (const int*)  d_in[2];
    const int*   right_len = (const int*)  d_in[3];
    const float* emb       = (const float*)d_in[4];
    const float* W_l       = (const float*)d_in[5];
    const float* b_l       = (const float*)d_in[6];
    const float* W_r       = (const float*)d_in[7];
    const float* b_r       = (const float*)d_in[8];
    const float* W_d       = (const float*)d_in[9];
    const float* b_d       = (const float*)d_in[10];
    float* out = (float*)d_out;

    // workspace: [ wb (2 sides, bf16 swizzled) | hcat f32[256][1024] ]
    __bf16* wb   = (__bf16*)d_ws;
    float*  hcat = (float*)((char*)d_ws + 2 * WB_SIDE * sizeof(__bf16));

    // 1) weight conversion + swizzle (re-done every call: deterministic)
    {
        int total = 2 * KPAD * GATES;
        prep_weights<<<(total + 255) / 256, 256, 0, stream>>>(W_l, W_r, wb);
    }
    // 2) fused twin-LSTM recurrence: 32 independent workgroups
    lstm_recur<<<32, 256, SM_TOT, stream>>>(left_ids, right_ids, left_len,
                                            right_len, emb, b_l, b_r, wb, hcat);
    // 3) dense head
    logits_kernel<<<(BATCH * NCLS + 255) / 256, 256, 0, stream>>>(hcat, W_d, b_d, out);
}
